// MultiHeadCrossAttentionWithCudaKernel_12171937317145
// MI455X (gfx1250) — 16-bit and 8-bit matrix operand order repaired and checked by reading, compiled, not run on hardware
//
#include <hip/hip_runtime.h>
#include <hip/hip_bf16.h>
#include <stdint.h>

typedef _Float16 v16h __attribute__((ext_vector_type(16)));
typedef _Float16 v8h  __attribute__((ext_vector_type(8)));
typedef float    v8f  __attribute__((ext_vector_type(8)));
typedef int      v8i  __attribute__((ext_vector_type(8)));
typedef int      v4i  __attribute__((ext_vector_type(4)));
typedef int      v2i  __attribute__((ext_vector_type(2)));

#if __has_builtin(__builtin_amdgcn_global_load_async_to_lds_b128) && \
    __has_builtin(__builtin_amdgcn_s_wait_asynccnt)
#define USE_ASYNC_LDS 1
#else
#define USE_ASYNC_LDS 0
#endif

#if USE_ASYNC_LDS
typedef int gvec4 __attribute__((vector_size(4 * sizeof(int))));
typedef __attribute__((address_space(1))) gvec4 gvec4_g;
typedef __attribute__((address_space(3))) gvec4 gvec4_l;
__device__ __forceinline__ void async_cp32(const _Float16* gsrc, _Float16* ldst) {
  gvec4_g* g = (gvec4_g*)(uintptr_t)gsrc;
  gvec4_l* l = (gvec4_l*)(uint32_t)(uintptr_t)ldst;
  __builtin_amdgcn_global_load_async_to_lds_b128(g, l, 0, 0);
  __builtin_amdgcn_global_load_async_to_lds_b128(g, l, 16, 0);
}
#endif

__device__ __forceinline__ v8i load_a_i8(const int8_t* p) {
  v2i a0 = *(const v2i*)(p);
  v2i a1 = *(const v2i*)(p + 16);
  v2i a2 = *(const v2i*)(p + 32);
  v2i a3 = *(const v2i*)(p + 48);
  v8i r = {a0.x, a0.y, a1.x, a1.y, a2.x, a2.y, a3.x, a3.y};
  return r;
}
__device__ __forceinline__ v8i load_b_i8(const int8_t* p) {
  v4i b0 = *(const v4i*)(p);
  v4i b1 = *(const v4i*)(p + 32);
  v8i r = {b0.x, b0.y, b0.z, b0.w, b1.x, b1.y, b1.z, b1.w};
  return r;
}
__device__ __forceinline__ v16h load_a_f16(const _Float16* p) {
  v8h lo = *(const v8h*)(p);
  v8h hi = *(const v8h*)(p + 16);
  return __builtin_shufflevector(lo, hi, 0, 1, 2, 3, 4, 5, 6, 7, 8, 9, 10, 11,
                                 12, 13, 14, 15);
}

template <typename TIN>
__global__ void quant_rows_kernel(const TIN* __restrict__ X, int8_t* __restrict__ Xq,
                                  float* __restrict__ sX, int K) {
  const int row = blockIdx.x;
  const int tid = threadIdx.x;
  const size_t base = (size_t)row * K;
  __shared__ float red[256];
  float m = 0.f;
  for (int i = tid; i < K; i += 256) m = fmaxf(m, fabsf((float)X[base + i]));
  red[tid] = m;
  __syncthreads();
  for (int s = 128; s > 0; s >>= 1) {
    if (tid < s) red[tid] = fmaxf(red[tid], red[tid + s]);
    __syncthreads();
  }
  const float scale = fmaxf(red[0] * (1.f / 127.f), 1e-8f);
  if (tid == 0) sX[row] = scale;
  const float inv = 1.f / scale;
  for (int i = tid; i < K; i += 256) {
    float q = rintf((float)X[base + i] * inv);
    q = fminf(fmaxf(q, -127.f), 127.f);
    Xq[base + i] = (int8_t)(int)q;
  }
}

__global__ void f32_to_f16_kernel(const float* __restrict__ in,
                                  _Float16* __restrict__ out, size_t n) {
  size_t i = (size_t)blockIdx.x * blockDim.x + threadIdx.x;
  if (i < n) out[i] = (_Float16)in[i];
}

template <bool OUT_F16>
__global__ void __launch_bounds__(128)
gemm_w8a8_kernel(const int8_t* __restrict__ Aq, const float* __restrict__ sA,
                 const int8_t* __restrict__ Bq, const float* __restrict__ sB,
                 const float* __restrict__ bias, void* __restrict__ Cout,
                 int T, int Nout, int K) {
  const int tid = threadIdx.x;
  const int wave = tid >> 5;
  const int lane = tid & 31;
  const int hl = lane >> 4;
  const int lm = lane & 15;
  const int row0 = blockIdx.x * 64 + (wave & 1) * 32;
  const int col0 = blockIdx.y * 128 + (wave >> 1) * 64;

  const int8_t* ap0 = Aq + (size_t)(row0 + lm) * K + hl * 8;
  const int8_t* ap1 = ap0 + (size_t)16 * K;
  const int8_t* bp0 = Bq + (size_t)(col0 + lm) * K + hl * 8;
  const size_t bstride = (size_t)16 * K;

  v8i acc[2][4];
  const v8i izero = {};
#pragma unroll
  for (int i = 0; i < 2; ++i)
#pragma unroll
    for (int j = 0; j < 4; ++j) acc[i][j] = izero;

  for (int kc = 0; kc < K; kc += 64) {
    __builtin_prefetch(ap0 + kc + 256, 0, 0);
    v8i a[2];
    a[0] = load_a_i8(ap0 + kc);
    a[1] = load_a_i8(ap1 + kc);
    v8i bf[4];
#pragma unroll
    for (int j = 0; j < 4; ++j) bf[j] = load_a_i8(bp0 + j * bstride + kc);
#pragma unroll
    for (int i = 0; i < 2; ++i)
#pragma unroll
      for (int j = 0; j < 4; ++j)
        acc[i][j] = __builtin_amdgcn_wmma_i32_16x16x64_iu8(
            true, a[i], true, bf[j], acc[i][j], false, false);
  }

  float sa[2][8];
#pragma unroll
  for (int i = 0; i < 2; ++i)
#pragma unroll
    for (int r = 0; r < 8; ++r) sa[i][r] = sA[row0 + i * 16 + r + hl * 8];

#pragma unroll
  for (int j = 0; j < 4; ++j) {
    const int col = col0 + j * 16 + lm;
    const float sb = sB[col];
    const float bi = bias[col];
#pragma unroll
    for (int i = 0; i < 2; ++i) {
#pragma unroll
      for (int r = 0; r < 8; ++r) {
        const int row = row0 + i * 16 + r + hl * 8;
        const float v = (float)acc[i][j][r] * sa[i][r] * sb + bi;
        if (OUT_F16)
          ((_Float16*)Cout)[(size_t)row * Nout + col] = (_Float16)v;
        else
          ((float*)Cout)[(size_t)row * Nout + col] = v;
      }
    }
  }
}

__global__ void __launch_bounds__(128)
gemm_f16_kernel(const _Float16* __restrict__ A, const _Float16* __restrict__ Wt,
                const float* __restrict__ bias, _Float16* __restrict__ C,
                int T, int Nout, int K) {
  const int tid = threadIdx.x;
  const int wave = tid >> 5;
  const int lane = tid & 31;
  const int hl = lane >> 4;
  const int lm = lane & 15;
  const int row0 = blockIdx.x * 64 + (wave & 1) * 32;
  const int col0 = blockIdx.y * 128 + (wave >> 1) * 64;

  const _Float16* ap0 = A + (size_t)(row0 + lm) * K + hl * 8;
  const _Float16* ap1 = ap0 + (size_t)16 * K;
  const _Float16* bp0 = Wt + (size_t)(col0 + lm) * K + hl * 8;
  const size_t bstride = (size_t)16 * K;

  v8f acc[2][4];
  const v8f fzero = {};
#pragma unroll
  for (int i = 0; i < 2; ++i)
#pragma unroll
    for (int j = 0; j < 4; ++j) acc[i][j] = fzero;

  for (int kc = 0; kc < K; kc += 32) {
    v16h a[2];
    a[0] = load_a_f16(ap0 + kc);
    a[1] = load_a_f16(ap1 + kc);
    v16h bf[4];
#pragma unroll
    for (int j = 0; j < 4; ++j) bf[j] = load_a_f16(bp0 + j * bstride + kc);
#pragma unroll
    for (int i = 0; i < 2; ++i)
#pragma unroll
      for (int j = 0; j < 4; ++j)
        acc[i][j] = __builtin_amdgcn_wmma_f32_16x16x32_f16(
            false, a[i], false, bf[j], (short)0, acc[i][j], false, false);
  }

#pragma unroll
  for (int j = 0; j < 4; ++j) {
    const int col = col0 + j * 16 + lm;
    const float bi = bias[col];
#pragma unroll
    for (int i = 0; i < 2; ++i) {
#pragma unroll
      for (int r = 0; r < 8; ++r) {
        const int row = row0 + i * 16 + r + hl * 8;
        C[(size_t)row * Nout + col] = (_Float16)(acc[i][j][r] + bi);
      }
    }
  }
}

__global__ void __launch_bounds__(256)
attn_kernel(const _Float16* __restrict__ Q, const _Float16* __restrict__ KV,
            _Float16* __restrict__ O) {
  constexpr int Cdim = 2048, KVC = 4096, Mkeys = 512, Dh = 128, Nq = 4096;
  constexpr int NB = Mkeys / 32;
  const int b = blockIdx.z, h = blockIdx.y;
  const int tid = threadIdx.x;
  const int wave = tid >> 5, lane = tid & 31, hl = lane >> 4, lm = lane & 15;
  const int q0 = blockIdx.x * 128 + wave * 16;

#if USE_ASYNC_LDS
  __shared__ _Float16 lds_k[2][32 * Dh];
  __shared__ _Float16 lds_v[2][32 * Dh];
#else
  __shared__ _Float16 lds_k[1][32 * Dh];
  __shared__ _Float16 lds_v[1][32 * Dh];
#endif

  const _Float16* qrow = Q + ((size_t)(b * Nq + q0 + lm)) * Cdim + h * Dh;
  v16h qb[4];
#pragma unroll
  for (int c = 0; c < 4; ++c) qb[c] = load_a_f16(qrow + c * 32 + hl * 8);

  v8f acc_o[8];
  const v8f vzero = {};
#pragma unroll
  for (int t = 0; t < 8; ++t) acc_o[t] = vzero;
  float mprev = -__builtin_inff(), lsum = 0.f;
  const float smul = 0.088388347648318447f;

  const _Float16* kbase_g = KV + (size_t)b * Mkeys * KVC + h * Dh;
  const _Float16* vbase_g = kbase_g + Cdim;

  const int idx = tid * 16;
  const int krow = idx >> 7;
  const int kcol = idx & 127;
  auto stage = [&](int kb, int buf) {
    const size_t goff = (size_t)(kb * 32 + krow) * KVC + kcol;
#if USE_ASYNC_LDS
    async_cp32(kbase_g + goff, &lds_k[buf][idx]);
    async_cp32(vbase_g + goff, &lds_v[buf][idx]);
#else
    *(v16h*)(&lds_k[buf][idx]) = *(const v16h*)(kbase_g + goff);
    *(v16h*)(&lds_v[buf][idx]) = *(const v16h*)(vbase_g + goff);
#endif
  };

#if USE_ASYNC_LDS
  stage(0, 0);
  __builtin_amdgcn_s_wait_asynccnt(0);
  __syncthreads();
#endif

  for (int kb = 0; kb < NB; ++kb) {
#if USE_ASYNC_LDS
    const int buf = kb & 1;
    if (kb + 1 < NB) stage(kb + 1, buf ^ 1);
#else
    const int buf = 0;
    stage(kb, 0);
    __syncthreads();
#endif
    const _Float16* lk = lds_k[buf];
    const _Float16* lv = lds_v[buf];

    v8f s0 = {}, s1 = {};
#pragma unroll
    for (int c = 0; c < 4; ++c) {
      v16h a0 = load_a_f16(lk + lm * Dh + c * 32 + hl * 8);
      v16h a1 = load_a_f16(lk + (16 + lm) * Dh + c * 32 + hl * 8);
      s0 = __builtin_amdgcn_wmma_f32_16x16x32_f16(false, a0, false, qb[c],
                                                  (short)0, s0, false, false);
      s1 = __builtin_amdgcn_wmma_f32_16x16x32_f16(false, a1, false, qb[c],
                                                  (short)0, s1, false, false);
    }

    float mloc = -__builtin_inff();
#pragma unroll
    for (int r = 0; r < 8; ++r) {
      s0[r] *= smul;
      s1[r] *= smul;
      mloc = fmaxf(mloc, fmaxf(s0[r], s1[r]));
    }
    const float mpart = __shfl_xor(mloc, 16);
    const float mnew = fmaxf(mprev, fmaxf(mloc, mpart));
    const float alpha = __expf(mprev - mnew);
    float p0[8], p1[8];
    float ssum = 0.f;
#pragma unroll
    for (int r = 0; r < 8; ++r) {
      p0[r] = __expf(s0[r] - mnew);
      p1[r] = __expf(s1[r] - mnew);
      ssum += p0[r] + p1[r];
    }
    ssum += __shfl_xor(ssum, 16);
    lsum = lsum * alpha + ssum;
    mprev = mnew;
#pragma unroll
    for (int t = 0; t < 8; ++t) {
#pragma unroll
      for (int r = 0; r < 8; ++r) acc_o[t][r] *= alpha;
    }

    v16h pfrag;
#pragma unroll
    for (int r = 0; r < 8; ++r) {
      const float x0 = __shfl_xor(p0[r], 16);
      const float x1 = __shfl_xor(p1[r], 16);
      const float lo = hl ? x1 : p0[r];
      const float hi = hl ? p1[r] : x0;
      pfrag[r] = (_Float16)lo;
      pfrag[r + 8] = (_Float16)hi;
    }

#pragma unroll
    for (int t = 0; t < 8; ++t) {
      const int d = t * 16 + lm;
      const int kk = hl * 8;
      v16h vfrag;
#pragma unroll
      for (int j = 0; j < 8; ++j) {
        vfrag[j] = lv[(2 * kk + j) * Dh + d];
        vfrag[j + 8] = lv[(2 * kk + 8 + j) * Dh + d];
      }
      acc_o[t] = __builtin_amdgcn_wmma_f32_16x16x32_f16(
          false, vfrag, false, pfrag, (short)0, acc_o[t], false, false);
    }

#if USE_ASYNC_LDS
    __builtin_amdgcn_s_wait_asynccnt(0);
#endif
    __syncthreads();
  }

  const float inv = 1.f / lsum;
  _Float16* orow = O + ((size_t)(b * Nq + q0 + lm)) * Cdim + h * Dh;
#pragma unroll
  for (int t = 0; t < 8; ++t) {
#pragma unroll
    for (int r = 0; r < 8; ++r)
      orow[t * 16 + r + hl * 8] = (_Float16)(acc_o[t][r] * inv);
  }
}

extern "C" void kernel_launch(void* const* d_in, const int* in_sizes, int n_in,
                              void* d_out, int out_size, void* d_ws, size_t ws_size,
                              hipStream_t stream) {
  const float* x    = (const float*)d_in[0];
  const float* cond = (const float*)d_in[1];
  const float* wq   = (const float*)d_in[2];
  const float* bq   = (const float*)d_in[3];
  const float* wkv  = (const float*)d_in[4];
  const float* bkv  = (const float*)d_in[5];
  const float* wp   = (const float*)d_in[6];
  const float* bp   = (const float*)d_in[7];
  float* out = (float*)d_out;
  (void)in_sizes; (void)n_in; (void)out_size; (void)ws_size;

  const int B = 2, N = 4096, M = 512, D = 2048, H = 16;
  const int T = B * N;
  const int TM = B * M;
  const int KVC = 2 * D;

  uint8_t* ws = (uint8_t*)d_ws;
  size_t off = 0;
  auto alloc = [&](size_t bytes) -> void* {
    void* p = ws + off;
    off = (off + bytes + 255) & ~(size_t)255;
    return p;
  };
  int8_t*   xq   = (int8_t*)  alloc((size_t)T * D);
  float*    sx   = (float*)   alloc((size_t)T * 4);
  int8_t*   wqq  = (int8_t*)  alloc((size_t)D * D);
  float*    swq  = (float*)   alloc((size_t)D * 4);
  int8_t*   wpq  = (int8_t*)  alloc((size_t)D * D);
  float*    swp  = (float*)   alloc((size_t)D * 4);
  _Float16* qf   = (_Float16*)alloc((size_t)T * D * 2);
  _Float16* cf   = (_Float16*)alloc((size_t)TM * D * 2);
  _Float16* wkvf = (_Float16*)alloc((size_t)KVC * D * 2);
  _Float16* kvf  = (_Float16*)alloc((size_t)TM * KVC * 2);
  _Float16* of   = (_Float16*)alloc((size_t)T * D * 2);
  int8_t*   oq   = (int8_t*)  alloc((size_t)T * D);
  float*    so   = (float*)   alloc((size_t)T * 4);

  quant_rows_kernel<float><<<T, 256, 0, stream>>>(x, xq, sx, D);
  quant_rows_kernel<float><<<D, 256, 0, stream>>>(wq, wqq, swq, D);
  quant_rows_kernel<float><<<D, 256, 0, stream>>>(wp, wpq, swp, D);
  {
    size_t n = (size_t)TM * D;
    f32_to_f16_kernel<<<(n + 255) / 256, 256, 0, stream>>>(cond, cf, n);
  }
  {
    size_t n = (size_t)KVC * D;
    f32_to_f16_kernel<<<(n + 255) / 256, 256, 0, stream>>>(wkv, wkvf, n);
  }
  gemm_w8a8_kernel<true><<<dim3(T / 64, D / 128), 128, 0, stream>>>(
      xq, sx, wqq, swq, bq, qf, T, D, D);
  gemm_f16_kernel<<<dim3(TM / 64, KVC / 128), 128, 0, stream>>>(
      cf, wkvf, bkv, kvf, TM, KVC, D);
  attn_kernel<<<dim3(N / 128, H, B), 256, 0, stream>>>(qf, kvf, of);
  quant_rows_kernel<_Float16><<<T, 256, 0, stream>>>(of, oq, so, D);
  gemm_w8a8_kernel<false><<<dim3(T / 64, D / 128), 128, 0, stream>>>(
      oq, so, wpq, swp, bp, out, T, D, D);
}
